// Decoder_27633819582887
// MI455X (gfx1250) — matrix operand order repaired and checked by reading, not compiled, not run on hardware
//
#include <hip/hip_runtime.h>
#include <hip/hip_bf16.h>

// ---------------- problem constants ----------------
static constexpr int NB   = 32768;  // batch
static constexpr int NEMB = 64;
static constexpr int NH   = 128;
static constexpr int NMLP = 1024;
static constexpr int NZ   = 32;
static constexpr int NZX  = NMLP + NZ;   // 1056
static constexpr int NSEQ = 12;
static constexpr int NG   = 4 * NH;      // 512 gate columns

typedef __attribute__((ext_vector_type(16))) _Float16 v16h;
typedef __attribute__((ext_vector_type(8)))  _Float16 v8h;
typedef __attribute__((ext_vector_type(8)))  float    v8f;

union V16HU { v16h v; v8h h[2]; };

// ---------------- WMMA fragment loaders (per CDNA5 ISA layouts) ----------------
// A (16x32 f16): lanes 0-15 hold row M=lane, K 0-7 (v0-3) and 16-23 (v4-7);
//                lanes 16-31 hold row M=lane-16, K 8-15 and 24-31.
__device__ __forceinline__ v16h load_a_frag(const _Float16* __restrict__ X, int ldx,
                                            int m0, int k0, int lane) {
  int m  = m0 + (lane & 15);
  int kb = (lane >> 4) << 3;                 // 0 or 8
  const _Float16* p = X + (size_t)m * ldx + k0 + kb;
  V16HU r;
  r.h[0] = *(const v8h*)(p);                 // K = k0+kb .. +7
  r.h[1] = *(const v8h*)(p + 16);            // K = k0+kb+16 .. +23
  return r.v;
}

// B (32x16 f16, element (k,n)=W[n][k] for out = X*W^T):
// lane holds column N=lane&15; lanes 0-15 K 0-7,16-23, lanes 16-31 K 8-15,24-31 (matches A).
__device__ __forceinline__ v16h load_b_frag(const _Float16* __restrict__ W, int ldw,
                                            int n0, int k0, int lane) {
  int n  = n0 + (lane & 15);
  int kb = (lane >> 4) << 3;                 // 0 or 8
  const _Float16* p = W + (size_t)n * ldw + k0 + kb;
  V16HU r;
  r.h[0] = *(const v8h*)(p);
  r.h[1] = *(const v8h*)(p + 16);
  return r.v;
}

// ---------------- generic GEMM: Y = act(X(MxK) * W(NxK)^T + bias) ----------------
// grid = (N/64, M/256); block = 256 (8 waves); wave tile = 32 rows x 64 cols.
__global__ void __launch_bounds__(256)
k_gemm_relu_f16(const _Float16* __restrict__ X, int ldx, int K,
                const _Float16* __restrict__ W,
                const float* __restrict__ bias,
                _Float16* __restrict__ Y, int N,
                float* __restrict__ czero) {
  const int lane = threadIdx.x & 31;
  const int wid  = threadIdx.x >> 5;
  const int m0   = (blockIdx.y * 8 + wid) * 32;
  const int n0   = blockIdx.x * 64;

  v8f acc[2][4];
  v8f vzero = {};
  #pragma unroll
  for (int mt = 0; mt < 2; ++mt)
    #pragma unroll
    for (int g = 0; g < 4; ++g) acc[mt][g] = vzero;

  for (int k0 = 0; k0 < K; k0 += 32) {
    if (k0 + 64 < K)  // speculative prefetch of a future A chunk (global_prefetch_b8)
      __builtin_prefetch(X + (size_t)(m0 + (lane & 15)) * ldx + k0 + 64, 0, 1);
    v16h a0 = load_a_frag(X, ldx, m0,      k0, lane);
    v16h a1 = load_a_frag(X, ldx, m0 + 16, k0, lane);
    #pragma unroll
    for (int g = 0; g < 4; ++g) {
      v16h b = load_b_frag(W, K, n0 + g * 16, k0, lane);
      acc[0][g] = __builtin_amdgcn_wmma_f32_16x16x32_f16(false, a0, false, b, (short)0, acc[0][g], false, false);
      acc[1][g] = __builtin_amdgcn_wmma_f32_16x16x32_f16(false, a1, false, b, (short)0, acc[1][g], false, false);
    }
  }

  const int nl = lane & 15;
  const int mo = (lane >> 4) << 3;           // C/D: lanes 16-31 hold M=r+8
  #pragma unroll
  for (int g = 0; g < 4; ++g) {
    int n = n0 + g * 16 + nl;
    float bv = bias[n];
    #pragma unroll
    for (int mt = 0; mt < 2; ++mt)
      #pragma unroll
      for (int r = 0; r < 8; ++r) {
        int m = m0 + mt * 16 + mo + r;
        float v = acc[mt][g][r] + bv;
        v = v > 0.f ? v : 0.f;
        Y[(size_t)m * N + n] = (_Float16)v;
        if (czero) czero[(size_t)m * N + n] = 0.f;   // zero LSTM c (GEMM2 only, N==NH)
      }
  }
}

// same tiling; f32 output, two biases summed, no relu (zx_proj)
__global__ void __launch_bounds__(256)
k_gemm_bias2_f32(const _Float16* __restrict__ X, int ldx, int K,
                 const _Float16* __restrict__ W,
                 const float* __restrict__ ba, const float* __restrict__ bb,
                 float* __restrict__ Y, int N) {
  const int lane = threadIdx.x & 31;
  const int wid  = threadIdx.x >> 5;
  const int m0   = (blockIdx.y * 8 + wid) * 32;
  const int n0   = blockIdx.x * 64;

  v8f acc[2][4];
  v8f vzero = {};
  #pragma unroll
  for (int mt = 0; mt < 2; ++mt)
    #pragma unroll
    for (int g = 0; g < 4; ++g) acc[mt][g] = vzero;

  for (int k0 = 0; k0 < K; k0 += 32) {
    if (k0 + 64 < K)
      __builtin_prefetch(X + (size_t)(m0 + (lane & 15)) * ldx + k0 + 64, 0, 1);
    v16h a0 = load_a_frag(X, ldx, m0,      k0, lane);
    v16h a1 = load_a_frag(X, ldx, m0 + 16, k0, lane);
    #pragma unroll
    for (int g = 0; g < 4; ++g) {
      v16h b = load_b_frag(W, K, n0 + g * 16, k0, lane);
      acc[0][g] = __builtin_amdgcn_wmma_f32_16x16x32_f16(false, a0, false, b, (short)0, acc[0][g], false, false);
      acc[1][g] = __builtin_amdgcn_wmma_f32_16x16x32_f16(false, a1, false, b, (short)0, acc[1][g], false, false);
    }
  }

  const int nl = lane & 15;
  const int mo = (lane >> 4) << 3;
  #pragma unroll
  for (int g = 0; g < 4; ++g) {
    int n = n0 + g * 16 + nl;
    float bv = ba[n] + bb[n];
    #pragma unroll
    for (int mt = 0; mt < 2; ++mt)
      #pragma unroll
      for (int r = 0; r < 8; ++r) {
        int m = m0 + mt * 16 + mo + r;
        Y[(size_t)m * N + n] = acc[mt][g][r] + bv;
      }
  }
}

// ---------------- fused LSTM step ----------------
// gates = zx_proj + [dec|h] * [W_emb|W_hh]^T  (K = 64+128 = 192)
// wave: 32 rows x 16 h-cols, computing that column slice in all four gates.
// grid = (NH/16, NB/256); block = 256.
__global__ void __launch_bounds__(256)
k_lstm_step(const _Float16* __restrict__ dec,
            const _Float16* __restrict__ hin,
            const _Float16* __restrict__ Wemb,   // 512 x 64  f16
            const _Float16* __restrict__ Whh,    // 512 x 128 f16
            const float* __restrict__ zproj,     // B x 512
            float* __restrict__ c,               // B x 128 (in place)
            _Float16* __restrict__ hout) {       // B x 128
  const int lane = threadIdx.x & 31;
  const int wid  = threadIdx.x >> 5;
  const int m0   = (blockIdx.y * 8 + wid) * 32;
  const int j0   = blockIdx.x * 16;

  const int nl = lane & 15;
  const int mo = (lane >> 4) << 3;

  // seed accumulators with zx_proj (biases already folded in)
  v8f acc[2][4];
  #pragma unroll
  for (int g = 0; g < 4; ++g) {
    int n = g * NH + j0 + nl;
    #pragma unroll
    for (int mt = 0; mt < 2; ++mt) {
      v8f a;
      #pragma unroll
      for (int r = 0; r < 8; ++r)
        a[r] = zproj[(size_t)(m0 + mt * 16 + mo + r) * NG + n];
      acc[mt][g] = a;
    }
  }

  #pragma unroll
  for (int kk = 0; kk < 6; ++kk) {      // K blocks: 0,32 from dec; 64..191 from h
    v16h a0, a1;
    if (kk < 2) {
      a0 = load_a_frag(dec, NEMB, m0,      kk * 32, lane);
      a1 = load_a_frag(dec, NEMB, m0 + 16, kk * 32, lane);
    } else {
      a0 = load_a_frag(hin, NH, m0,      kk * 32 - 64, lane);
      a1 = load_a_frag(hin, NH, m0 + 16, kk * 32 - 64, lane);
    }
    #pragma unroll
    for (int g = 0; g < 4; ++g) {
      int n0 = g * NH + j0;
      v16h b = (kk < 2) ? load_b_frag(Wemb, NEMB, n0, kk * 32, lane)
                        : load_b_frag(Whh,  NH,   n0, kk * 32 - 64, lane);
      acc[0][g] = __builtin_amdgcn_wmma_f32_16x16x32_f16(false, a0, false, b, (short)0, acc[0][g], false, false);
      acc[1][g] = __builtin_amdgcn_wmma_f32_16x16x32_f16(false, a1, false, b, (short)0, acc[1][g], false, false);
    }
  }

  // LSTM cell update
  #pragma unroll
  for (int mt = 0; mt < 2; ++mt)
    #pragma unroll
    for (int r = 0; r < 8; ++r) {
      int m = m0 + mt * 16 + mo + r;
      size_t idx = (size_t)m * NH + j0 + nl;
      float iv = acc[mt][0][r], fv = acc[mt][1][r];
      float gv = acc[mt][2][r], ov = acc[mt][3][r];
      float si = 1.f / (1.f + __expf(-iv));
      float sf = 1.f / (1.f + __expf(-fv));
      float so = 1.f / (1.f + __expf(-ov));
      float tg = tanhf(gv);
      float cn = sf * c[idx] + si * tg;
      c[idx] = cn;
      hout[idx] = (_Float16)(so * tanhf(cn));
    }
}

// ---------------- per-step position head: rel_pos -> pred slice + next dec_inp ----------------
__global__ void __launch_bounds__(256)
k_pos(const _Float16* __restrict__ h, const float* __restrict__ Whp,
      const float* __restrict__ bhp, const float* __restrict__ Wsp,
      const float* __restrict__ bsp, float* __restrict__ pred_t,
      _Float16* __restrict__ dec) {
  __shared__ float sWhp[2 * NH];
  __shared__ float sWsp[NEMB * 2];
  __shared__ float sbsp[NEMB];
  int t = threadIdx.x;
  if (t < 2 * NH)   sWhp[t] = Whp[t];
  if (t < NEMB * 2) sWsp[t] = Wsp[t];
  if (t < NEMB)     sbsp[t] = bsp[t];
  __syncthreads();

  int b = blockIdx.x * blockDim.x + t;
  float r0 = bhp[0], r1 = bhp[1];
  const v8h* hv = (const v8h*)(h + (size_t)b * NH);
  #pragma unroll
  for (int q = 0; q < 16; ++q) {
    v8h x = hv[q];
    #pragma unroll
    for (int j = 0; j < 8; ++j) {
      float hvf = (float)x[j];
      int k = q * 8 + j;
      r0 += hvf * sWhp[k];
      r1 += hvf * sWhp[NH + k];
    }
  }
  pred_t[(size_t)b * 2]     = r0;
  pred_t[(size_t)b * 2 + 1] = r1;

  _Float16* dp = dec + (size_t)b * NEMB;
  #pragma unroll
  for (int q = 0; q < 8; ++q) {
    v8h o;
    #pragma unroll
    for (int j = 0; j < 8; ++j) {
      int e = q * 8 + j;
      o[j] = (_Float16)(r0 * sWsp[e * 2] + r1 * sWsp[e * 2 + 1] + sbsp[e]);
    }
    *(v8h*)(dp + q * 8) = o;
  }
}

// initial dec_inp from last_pos_rel
__global__ void __launch_bounds__(256)
k_dec_init(const float* __restrict__ lpr, const float* __restrict__ Wsp,
           const float* __restrict__ bsp, _Float16* __restrict__ dec) {
  __shared__ float sWsp[NEMB * 2];
  __shared__ float sbsp[NEMB];
  int t = threadIdx.x;
  if (t < NEMB * 2) sWsp[t] = Wsp[t];
  if (t < NEMB)     sbsp[t] = bsp[t];
  __syncthreads();
  int b = blockIdx.x * blockDim.x + t;
  float r0 = lpr[(size_t)b * 2], r1 = lpr[(size_t)b * 2 + 1];
  _Float16* dp = dec + (size_t)b * NEMB;
  #pragma unroll
  for (int q = 0; q < 8; ++q) {
    v8h o;
    #pragma unroll
    for (int j = 0; j < 8; ++j) {
      int e = q * 8 + j;
      o[j] = (_Float16)(r0 * sWsp[e * 2] + r1 * sWsp[e * 2 + 1] + sbsp[e]);
    }
    *(v8h*)(dp + q * 8) = o;
  }
}

// zx = concat(enc_h_feat, z) as f16
__global__ void __launch_bounds__(256)
k_build_zx(const float* __restrict__ enc, const float* __restrict__ z,
           _Float16* __restrict__ zx) {
  size_t i = (size_t)blockIdx.x * blockDim.x + threadIdx.x;
  int b = (int)(i / NZX);
  int cidx = (int)(i % NZX);
  float v = (cidx < NMLP) ? enc[(size_t)b * NMLP + cidx]
                          : z[(size_t)b * NZ + (cidx - NMLP)];
  zx[i] = (_Float16)v;
}

// f32 submatrix -> packed f16
__global__ void __launch_bounds__(256)
k_cvt(const float* __restrict__ src, int ld, _Float16* __restrict__ dst,
      int cols, int total) {
  int i = blockIdx.x * blockDim.x + threadIdx.x;
  if (i >= total) return;
  int r = i / cols, cc = i % cols;
  dst[i] = (_Float16)src[(size_t)r * ld + cc];
}

// ---------------- host orchestration ----------------
extern "C" void kernel_launch(void* const* d_in, const int* in_sizes, int n_in,
                              void* d_out, int out_size, void* d_ws, size_t ws_size,
                              hipStream_t stream) {
  (void)in_sizes; (void)n_in; (void)out_size; (void)ws_size;
  const float* last_pos_rel = (const float*)d_in[1];
  const float* enc  = (const float*)d_in[2];
  const float* z    = (const float*)d_in[3];
  const float* W_ih = (const float*)d_in[5];   // 512 x 1120
  const float* W_hh = (const float*)d_in[6];   // 512 x 128
  const float* b_ih = (const float*)d_in[7];
  const float* b_hh = (const float*)d_in[8];
  const float* W1   = (const float*)d_in[9];   // 1024 x 1056
  const float* b1   = (const float*)d_in[10];
  const float* W2   = (const float*)d_in[11];  // 128 x 1024
  const float* b2   = (const float*)d_in[12];
  const float* W_sp = (const float*)d_in[13];  // 64 x 2
  const float* b_sp = (const float*)d_in[14];
  const float* W_hp = (const float*)d_in[15];  // 2 x 128
  const float* b_hp = (const float*)d_in[16];
  float* out = (float*)d_out;

  char* ws = (char*)d_ws;
  size_t off = 0;
  auto alloc = [&](size_t bytes) {
    off = (off + 255) & ~(size_t)255;
    void* p = ws + off;
    off += bytes;
    return p;
  };
  _Float16* zx    = (_Float16*)alloc((size_t)NB * NZX * 2);
  _Float16* hid   = (_Float16*)alloc((size_t)NB * NMLP * 2);
  float*    zproj = (float*)   alloc((size_t)NB * NG * 4);
  _Float16* hA    = (_Float16*)alloc((size_t)NB * NH * 2);
  _Float16* hB    = (_Float16*)alloc((size_t)NB * NH * 2);
  float*    cbuf  = (float*)   alloc((size_t)NB * NH * 4);
  _Float16* dec   = (_Float16*)alloc((size_t)NB * NEMB * 2);
  _Float16* w1h   = (_Float16*)alloc((size_t)NMLP * NZX * 2);
  _Float16* wzxh  = (_Float16*)alloc((size_t)NG * NZX * 2);
  _Float16* wembh = (_Float16*)alloc((size_t)NG * NEMB * 2);
  _Float16* whhh  = (_Float16*)alloc((size_t)NG * NH * 2);
  _Float16* w2h   = (_Float16*)alloc((size_t)NH * NMLP * 2);

  auto cvt = [&](const float* s, int ld, _Float16* d, int rows, int cols) {
    int total = rows * cols;
    k_cvt<<<(total + 255) / 256, 256, 0, stream>>>(s, ld, d, cols, total);
  };
  cvt(W1,          NZX,       w1h,   NMLP, NZX);   // W1
  cvt(W_ih,        NZX + NEMB, wzxh, NG,   NZX);   // W_zx = W_ih[:, :1056]
  cvt(W_ih + NZX,  NZX + NEMB, wembh, NG,  NEMB);  // W_emb = W_ih[:, 1056:]
  cvt(W_hh,        NH,        whhh,  NG,   NH);
  cvt(W2,          NMLP,      w2h,   NH,   NMLP);

  k_build_zx<<<(NB * NZX) / 256, 256, 0, stream>>>(enc, z, zx);
  k_dec_init<<<NB / 256, 256, 0, stream>>>(last_pos_rel, W_sp, b_sp, dec);

  // hid = relu(zx @ W1^T + b1)
  k_gemm_relu_f16<<<dim3(NMLP / 64, NB / 256), 256, 0, stream>>>(
      zx, NZX, NZX, w1h, b1, hid, NMLP, nullptr);
  // h0 = relu(hid @ W2^T + b2); also zero c
  k_gemm_relu_f16<<<dim3(NH / 64, NB / 256), 256, 0, stream>>>(
      hid, NMLP, NMLP, w2h, b2, hA, NH, cbuf);
  // zx_proj = zx @ W_zx^T + b_ih + b_hh
  k_gemm_bias2_f32<<<dim3(NG / 64, NB / 256), 256, 0, stream>>>(
      zx, NZX, NZX, wzxh, b_ih, b_hh, zproj, NG);

  _Float16* hin = hA;
  _Float16* ho  = hB;
  for (int t = 0; t < NSEQ; ++t) {
    k_lstm_step<<<dim3(NH / 16, NB / 256), 256, 0, stream>>>(
        dec, hin, wembh, whhh, zproj, cbuf, ho);
    k_pos<<<NB / 256, 256, 0, stream>>>(
        ho, W_hp, b_hp, W_sp, b_sp, out + (size_t)t * NB * 2, dec);
    _Float16* tmp = hin; hin = ho; ho = tmp;
  }
}
